// DecoderModel_72112500900063
// MI455X (gfx1250) — matrix operand order repaired and checked by reading, not compiled, not run on hardware
//
#include <hip/hip_runtime.h>
#include <hip/hip_bf16.h>
#include <math.h>

// ---------------------------------------------------------------------------
// MI455X (gfx1250) transformer decoder. All GEMMs + attention matmuls run on
// v_wmma_f32_16x16x32_bf16 (wave32). Residual stream kept in fp32.
// Round 3: fix async-to-LDS builtin pointer types (i32x4* with addrspace).
// ---------------------------------------------------------------------------

typedef __attribute__((ext_vector_type(16))) __bf16 bf16x16;
typedef __attribute__((ext_vector_type(8)))  __bf16 bf16x8;
typedef __attribute__((ext_vector_type(4)))  __bf16 bf16x4;
typedef __attribute__((ext_vector_type(2)))  __bf16 bf16x2;
typedef __attribute__((ext_vector_type(8)))  float  f32x8;
typedef __attribute__((ext_vector_type(4)))  unsigned int u32x4;
typedef __attribute__((ext_vector_type(8)))  int i32x8;
typedef __attribute__((ext_vector_type(4)))  int i32x4;

#define CAT16(lo, hi) __builtin_shufflevector(lo, hi, 0,1,2,3,4,5,6,7,8,9,10,11,12,13,14,15)
#define GLOBAL_AS __attribute__((address_space(1)))
#define LDS_AS    __attribute__((address_space(3)))

#if defined(__HIP_DEVICE_COMPILE__) && defined(__has_builtin)
#  if __has_builtin(__builtin_amdgcn_global_load_async_to_lds_b128) && \
      __has_builtin(__builtin_amdgcn_s_wait_asynccnt)
#    define USE_ASYNC_LDS 1
#  endif
#  if __has_builtin(__builtin_amdgcn_tensor_load_to_lds) && \
      __has_builtin(__builtin_amdgcn_s_wait_tensorcnt)
#    define USE_TDM 1
#  endif
#endif

constexpr int Bc  = 2;
constexpr int Tc  = 1024;
constexpr int Dc  = 1024;
constexpr int Hc  = 16;
constexpr int HDc = 64;
constexpr int FFc = 4096;
constexpr int Lc  = 4;
constexpr int MR  = Bc * Tc;          // 2048 rows in every activation matrix
constexpr float kAttnScale = 0.125f;  // HD^-0.5

// ---------------------------------------------------------------------------
// fp32 -> bf16 conversion (weights / activations feeding WMMA)
// ---------------------------------------------------------------------------
__global__ void k_f2b(const float* __restrict__ s, __bf16* __restrict__ d, long n) {
    long i = ((long)blockIdx.x * 256 + threadIdx.x) * 4;
    if (i + 3 < n) {
        d[i + 0] = (__bf16)s[i + 0];
        d[i + 1] = (__bf16)s[i + 1];
        d[i + 2] = (__bf16)s[i + 2];
        d[i + 3] = (__bf16)s[i + 3];
    }
}

// ---------------------------------------------------------------------------
// GEMM: C[M,N] = A[M,K](bf16) * W[K,N](bf16) + bias (+ residual), fp32 out.
// Workgroup tile 128x128, 8 waves, wave tile 32x64, K-step 32.
// Double-buffered LDS; A tile staged with async-to-LDS when available.
// ---------------------------------------------------------------------------
__launch_bounds__(256)
__global__ void k_gemm_bf16(const __bf16* __restrict__ A, const __bf16* __restrict__ W,
                            const float* __restrict__ bias, const float* __restrict__ residual,
                            float* __restrict__ C, int M, int N, int K) {
    __shared__ __bf16 As[2][128][40];    // 80B row stride: 16B aligned, conflict-free
    __shared__ __bf16 BsT[2][128][40];   // W tile stored transposed: BsT[n][k]

    const int tid  = threadIdx.x;
    const int lane = tid & 31, wid = tid >> 5;
    const int r4 = lane & 15, hf = lane >> 4;
    const int wm = wid >> 1, wn = wid & 1;
    const int bm = blockIdx.y * 128, bn = blockIdx.x * 128;

    f32x8 acc[2][4];
#pragma unroll
    for (int i = 0; i < 2; ++i)
#pragma unroll
        for (int j = 0; j < 4; ++j)
#pragma unroll
            for (int p = 0; p < 8; ++p) acc[i][j][p] = 0.f;

    const int ar = tid >> 1, ah = tid & 1;   // A staging: 128 rows x two 16-elem halves
    const int rp = tid >> 4, cg = tid & 15;  // W staging: 16 k-row-pairs x 16 col-groups(8)

    auto stage = [&](int buf, int k0) {
        // ---- A tile (128 x 32), row-major ----
        const __bf16* asrc = A + (size_t)(bm + ar) * K + k0 + ah * 16;
        __bf16* adst = &As[buf][ar][ah * 16];
#if defined(USE_ASYNC_LDS)
        __builtin_amdgcn_global_load_async_to_lds_b128(
            (GLOBAL_AS i32x4*)(asrc), (LDS_AS i32x4*)(adst), 0, 0);
        __builtin_amdgcn_global_load_async_to_lds_b128(
            (GLOBAL_AS i32x4*)(asrc + 8), (LDS_AS i32x4*)(adst + 8), 0, 0);
#else
        bf16x8 a0 = *(const bf16x8*)(asrc);
        bf16x8 a1 = *(const bf16x8*)(asrc + 8);
        *(bf16x8*)adst       = a0;
        *(bf16x8*)(adst + 8) = a1;
#endif
        // ---- W tile (32 x 128) -> transposed LDS; bf16 pairs fuse to b32 ----
        const __bf16* wsrc = W + (size_t)(k0 + 2 * rp) * N + bn + cg * 8;
        bf16x8 r0 = *(const bf16x8*)(wsrc);
        bf16x8 r1 = *(const bf16x8*)(wsrc + N);
#pragma unroll
        for (int e = 0; e < 8; ++e) {
            bf16x2 pr = {r0[e], r1[e]};
            *(bf16x2*)&BsT[buf][cg * 8 + e][2 * rp] = pr;   // k-pair contiguous, 4B aligned
        }
        if (k0 + 32 < K)   // gfx1250 global_prefetch_b8 two tiles ahead of compute
            __builtin_prefetch(W + (size_t)(k0 + 32 + 2 * rp) * N + bn + cg * 8, 0, 1);
    };

    const int nk = K >> 5;
    stage(0, 0);
    for (int it = 0; it < nk; ++it) {
        const int cur = it & 1;
#if defined(USE_ASYNC_LDS)
        __builtin_amdgcn_s_wait_asynccnt(0);
#endif
        __syncthreads();                      // tile `cur` ready for everyone
        if (it + 1 < nk) stage(cur ^ 1, (it + 1) << 5);   // overlap with compute

        bf16x16 afrag[2], bfrag[4];
#pragma unroll
        for (int i = 0; i < 2; ++i) {   // A-frag: lane holds row r4, K {8hf..} and {16+8hf..}
            int row = wm * 32 + i * 16 + r4;
            bf16x8 lo = *(const bf16x8*)&As[cur][row][8 * hf];
            bf16x8 hi = *(const bf16x8*)&As[cur][row][16 + 8 * hf];
            afrag[i] = CAT16(lo, hi);
        }
#pragma unroll
        for (int j = 0; j < 4; ++j) {   // B-frag: lane holds column r4, K {8hf..} and {16+8hf..}
            int col = wn * 64 + j * 16 + r4;
            bf16x8 lo = *(const bf16x8*)&BsT[cur][col][8 * hf];
            bf16x8 hi = *(const bf16x8*)&BsT[cur][col][16 + 8 * hf];
            bfrag[j] = CAT16(lo, hi);
        }
#pragma unroll
        for (int i = 0; i < 2; ++i)
#pragma unroll
            for (int j = 0; j < 4; ++j)
                acc[i][j] = __builtin_amdgcn_wmma_f32_16x16x32_bf16(
                    false, afrag[i], false, bfrag[j], (short)0, acc[i][j], false, false);
    }

#pragma unroll
    for (int i = 0; i < 2; ++i) {
        int row0 = bm + wm * 32 + i * 16 + 8 * hf;   // C layout: lanes 16-31 are rows +8
#pragma unroll
        for (int j = 0; j < 4; ++j) {
            int col = bn + wn * 64 + j * 16 + r4;
            float bc = bias ? bias[col] : 0.f;
#pragma unroll
            for (int p = 0; p < 8; ++p) {
                size_t idx = (size_t)(row0 + p) * N + col;
                float vv = acc[i][j][p] + bc;
                if (residual) vv += residual[idx];
                C[idx] = vv;
            }
        }
    }
}

// ---------------------------------------------------------------------------
// Flash attention. grid = B*H*(T/128), 256 threads (8 waves x 16 q-rows).
// q/k/v bf16 [B,H,T,HD]. Output bf16 packed to [B,T,D].
// K chunk staged by the Tensor Data Mover when available.
// ---------------------------------------------------------------------------
__launch_bounds__(256)
__global__ void k_attn(const __bf16* __restrict__ q, const __bf16* __restrict__ k,
                       const __bf16* __restrict__ v, const float* __restrict__ rb,
                       __bf16* __restrict__ ao) {
    __shared__ __bf16 Ks[32][72];        // K chunk, row-major [k][d], 144B stride
    __shared__ __bf16 VsT[64][40];       // V chunk transposed [d][k], 80B stride
    __shared__ __bf16 Ps[8][16][40];     // per-wave P round-trip (C layout -> A layout)

    const int tid = threadIdx.x, lane = tid & 31, wid = tid >> 5;
    const int r4 = lane & 15, hf = lane >> 4;
    const int qt = blockIdx.x & 7;        // T/128 = 8
    const int bh = blockIdx.x >> 3;       // b*H + h
    const int hh = bh & 15, bb = bh >> 4;
    const int qrow0 = qt * 128 + wid * 16;

    // Q A-fragments straight from global (rows are 128B contiguous)
    const __bf16* qbase = q + ((size_t)bh * Tc + qrow0 + r4) * HDc;
    bf16x16 aq[2];
#pragma unroll
    for (int i = 0; i < 2; ++i) {
        bf16x8 lo = *(const bf16x8*)(qbase + i * 32 + 8 * hf);
        bf16x8 hi = *(const bf16x8*)(qbase + i * 32 + 16 + 8 * hf);
        aq[i] = CAT16(lo, hi);
    }

    float mrow[8], lrow[8];
    f32x8 o[4];
#pragma unroll
    for (int p = 0; p < 8; ++p) { mrow[p] = -1e30f; lrow[p] = 0.f; }
#pragma unroll
    for (int j = 0; j < 4; ++j)
#pragma unroll
        for (int p = 0; p < 8; ++p) o[j][p] = 0.f;

#if defined(USE_TDM)
    const unsigned ldsK = (unsigned)(size_t)(LDS_AS void*)&Ks[0][0];
#endif
    const int rpv = tid >> 4, cgv = tid & 15;  // V staging: 16 k-row-pairs x 16 col-groups(4)

    for (int kb = 0; kb < Tc; kb += 32) {
#if defined(USE_TDM)
        if (wid == 0) {
            // TDM D#: 2D tile 64(d) x 32(k) of 2B elems, row-major, LDS pad
            // 4 DWORDs every 32 DWORDs -> 144B row stride in LDS.
            const unsigned long long ga =
                (unsigned long long)(size_t)(k + ((size_t)bh * Tc + kb) * HDc);
            u32x4 g0 = {1u,                                   // count=1, load descriptor
                        ldsK,                                 // lds_addr (bytes)
                        (unsigned)(ga & 0xFFFFFFFFu),
                        (unsigned)((ga >> 32) & 0x1FFFFFFu) | (2u << 30)};  // type=2
            i32x8 g1 = {(int)((1u << 16) | (1u << 20) | (4u << 22) | (3u << 25)),
                        (int)(64u << 16),          // tensor_dim0 = 64 (lo16 in dw1 hi)
                        (int)(32u << 16),          // tensor_dim0 hi=0 | tensor_dim1=32
                        (int)(64u << 16),          // tensor_dim1 hi=0 | tile_dim0=64
                        (int)(32u),                // tile_dim1=32, tile_dim2=0
                        (int)(64u),                // tensor_dim0_stride = 64 elems
                        0, 0};
            i32x4 z4 = {0, 0, 0, 0};
#if __clang_major__ >= 23
            i32x8 z8 = {0, 0, 0, 0, 0, 0, 0, 0};
            __builtin_amdgcn_tensor_load_to_lds(g0, g1, z4, z4, z8, 0);
#else
            __builtin_amdgcn_tensor_load_to_lds(g0, g1, z4, z4, 0);
#endif
            __builtin_amdgcn_s_wait_tensorcnt(0);
        }
#else
        {   // manual K staging: 32 rows x 8 groups of 8 elems
            const int kr = tid >> 3, g = tid & 7;
            const __bf16* ksrc = k + ((size_t)bh * Tc + kb + kr) * HDc + g * 8;
            *(bf16x8*)&Ks[kr][g * 8] = *(const bf16x8*)ksrc;
        }
#endif
        {   // V chunk transposed: bf16 k-pairs fuse to b32 stores
            const __bf16* vsrc = v + ((size_t)bh * Tc + kb + 2 * rpv) * HDc + cgv * 4;
            bf16x4 v0 = *(const bf16x4*)(vsrc);
            bf16x4 v1 = *(const bf16x4*)(vsrc + HDc);
#pragma unroll
            for (int e = 0; e < 4; ++e) {
                bf16x2 pr = {v0[e], v1[e]};
                *(bf16x2*)&VsT[cgv * 4 + e][2 * rpv] = pr;
            }
        }
        __syncthreads();

        // S = Q * K^T  (two 16-wide k tiles, contraction HD=64 in two WMMA steps)
        f32x8 s[2];
#pragma unroll
        for (int kc = 0; kc < 2; ++kc) {
            f32x8 av;
#pragma unroll
            for (int p = 0; p < 8; ++p) av[p] = 0.f;
#pragma unroll
            for (int dh = 0; dh < 2; ++dh) {
                bf16x8 lo = *(const bf16x8*)&Ks[kc * 16 + r4][dh * 32 + 8 * hf];
                bf16x8 hi = *(const bf16x8*)&Ks[kc * 16 + r4][dh * 32 + 16 + 8 * hf];
                bf16x16 bk = CAT16(lo, hi);
                av = __builtin_amdgcn_wmma_f32_16x16x32_bf16(
                    false, aq[dh], false, bk, (short)0, av, false, false);
            }
            s[kc] = av;
        }

        // online softmax; row = (p, hf) pair; reduce across the 16 lanes of one half
#pragma unroll
        for (int p = 0; p < 8; ++p) {
            int qg = qrow0 + 8 * hf + p;
            int kg0 = kb + r4, kg1 = kb + 16 + r4;
            int rel0 = min(max(kg0 - qg, -128), 128) + 128;
            int rel1 = min(max(kg1 - qg, -128), 128) + 128;
            float l0 = s[0][p] * kAttnScale + rb[rel0 * Hc + hh];
            float l1 = s[1][p] * kAttnScale + rb[rel1 * Hc + hh];
            float mx = fmaxf(l0, l1);
#pragma unroll
            for (int msk = 1; msk < 16; msk <<= 1) mx = fmaxf(mx, __shfl_xor(mx, msk, 32));
            float mn = fmaxf(mrow[p], mx);
            float al = __expf(mrow[p] - mn);
            mrow[p] = mn;
            float p0 = __expf(l0 - mn), p1 = __expf(l1 - mn);
            float rs = p0 + p1;
#pragma unroll
            for (int msk = 1; msk < 16; msk <<= 1) rs += __shfl_xor(rs, msk, 32);
            lrow[p] = lrow[p] * al + rs;
            s[0][p] = p0; s[1][p] = p1;
#pragma unroll
            for (int j = 0; j < 4; ++j) o[j][p] *= al;
        }

        // C-layout P -> LDS -> A-layout fragment
#pragma unroll
        for (int p = 0; p < 8; ++p) {
            int prow = p + 8 * hf;
            Ps[wid][prow][r4]      = (__bf16)s[0][p];
            Ps[wid][prow][16 + r4] = (__bf16)s[1][p];
        }
        __syncthreads();
        bf16x16 pa;
        {
            bf16x8 lo = *(const bf16x8*)&Ps[wid][r4][8 * hf];
            bf16x8 hi = *(const bf16x8*)&Ps[wid][r4][16 + 8 * hf];
            pa = CAT16(lo, hi);
        }
        // O += P * V
#pragma unroll
        for (int j = 0; j < 4; ++j) {
            bf16x8 lo = *(const bf16x8*)&VsT[j * 16 + r4][8 * hf];
            bf16x8 hi = *(const bf16x8*)&VsT[j * 16 + r4][16 + 8 * hf];
            bf16x16 bv = CAT16(lo, hi);
            o[j] = __builtin_amdgcn_wmma_f32_16x16x32_bf16(
                false, pa, false, bv, (short)0, o[j], false, false);
        }
        __syncthreads();
    }

#pragma unroll
    for (int p = 0; p < 8; ++p) {
        float inv = 1.f / lrow[p];
        int t = qrow0 + 8 * hf + p;
#pragma unroll
        for (int j = 0; j < 4; ++j) {
            int d = j * 16 + r4;
            ao[((size_t)bb * Tc + t) * Dc + hh * HDc + d] = (__bf16)(o[j][p] * inv);
        }
    }
}

// ---------------------------------------------------------------------------
// LayerNorm kernels (one block per row, D=1024, 256 threads x 4 elems)
// ---------------------------------------------------------------------------
__launch_bounds__(256)
__global__ void k_add_ln_f32(const float* __restrict__ t0, const float* __restrict__ ctx,
                             const float* __restrict__ g, const float* __restrict__ b,
                             float* __restrict__ h) {
    __shared__ float red[256];
    const int row = blockIdx.x, tid = threadIdx.x;
    float v[4]; float s = 0.f;
#pragma unroll
    for (int i = 0; i < 4; ++i) {
        int c = tid + i * 256;
        v[i] = t0[(size_t)row * Dc + c] + ctx[(size_t)row * Dc + c];
        s += v[i];
    }
    red[tid] = s; __syncthreads();
    for (int st = 128; st > 0; st >>= 1) { if (tid < st) red[tid] += red[tid + st]; __syncthreads(); }
    float mean = red[0] * (1.f / Dc); __syncthreads();
    s = 0.f;
#pragma unroll
    for (int i = 0; i < 4; ++i) { float d = v[i] - mean; s += d * d; }
    red[tid] = s; __syncthreads();
    for (int st = 128; st > 0; st >>= 1) { if (tid < st) red[tid] += red[tid + st]; __syncthreads(); }
    float rstd = rsqrtf(red[0] * (1.f / Dc) + 1e-5f);
#pragma unroll
    for (int i = 0; i < 4; ++i) {
        int c = tid + i * 256;
        h[(size_t)row * Dc + c] = (v[i] - mean) * rstd * g[c] + b[c];
    }
}

__launch_bounds__(256)
__global__ void k_ln_bf16(const float* __restrict__ x, const float* __restrict__ g,
                          const float* __restrict__ b, __bf16* __restrict__ out) {
    __shared__ float red[256];
    const int row = blockIdx.x, tid = threadIdx.x;
    float v[4]; float s = 0.f;
#pragma unroll
    for (int i = 0; i < 4; ++i) { v[i] = x[(size_t)row * Dc + tid + i * 256]; s += v[i]; }
    red[tid] = s; __syncthreads();
    for (int st = 128; st > 0; st >>= 1) { if (tid < st) red[tid] += red[tid + st]; __syncthreads(); }
    float mean = red[0] * (1.f / Dc); __syncthreads();
    s = 0.f;
#pragma unroll
    for (int i = 0; i < 4; ++i) { float d = v[i] - mean; s += d * d; }
    red[tid] = s; __syncthreads();
    for (int st = 128; st > 0; st >>= 1) { if (tid < st) red[tid] += red[tid + st]; __syncthreads(); }
    float rstd = rsqrtf(red[0] * (1.f / Dc) + 1e-5f);
#pragma unroll
    for (int i = 0; i < 4; ++i) {
        int c = tid + i * 256;
        out[(size_t)row * Dc + c] = (__bf16)((v[i] - mean) * rstd * g[c] + b[c]);
    }
}

// ---------------------------------------------------------------------------
// RoPE + head packing: qkv fp32 [B*T, 3*D] -> q/k/v bf16 [B,H,T,HD]
// ---------------------------------------------------------------------------
__global__ void k_rope_pack(const float* __restrict__ qkv, __bf16* __restrict__ qo,
                            __bf16* __restrict__ ko, __bf16* __restrict__ vo) {
    const long id = (long)blockIdx.x * 256 + threadIdx.x;   // B*T*H*32 threads
    const int i  = id & 31;
    const int hh = (id >> 5) & 15;
    const int t  = (id >> 9) & 1023;
    const int bb = (int)(id >> 19);
    const int d0 = 2 * i;
    const int fi = d0 & 31;                                  // freq index = (2i) % 32
    const float th = (float)t * __expf(-0.2878231366242557f * (float)fi); // 10000^(-fi/32)
    const float cs = __cosf(th), sn = __sinf(th);
    const size_t base = ((size_t)bb * Tc + t) * (3 * Dc) + hh * HDc + d0;
    const size_t obase = (((size_t)bb * Hc + hh) * Tc + t) * HDc + d0;
    {   // q with rope
        float x1 = qkv[base], x2 = qkv[base + 1];
        qo[obase]     = (__bf16)(x1 * cs - x2 * sn);
        qo[obase + 1] = (__bf16)(x1 * sn + x2 * cs);
    }
    {   // k with rope
        float x1 = qkv[base + Dc], x2 = qkv[base + Dc + 1];
        ko[obase]     = (__bf16)(x1 * cs - x2 * sn);
        ko[obase + 1] = (__bf16)(x1 * sn + x2 * cs);
    }
    {   // v copy
        vo[obase]     = (__bf16)qkv[base + 2 * Dc];
        vo[obase + 1] = (__bf16)qkv[base + 2 * Dc + 1];
    }
}

// ---------------------------------------------------------------------------
// Gated exact-GELU: gg[m,j] = bf16( ab[m,j] * gelu(ab[m,FF+j]) )
// ---------------------------------------------------------------------------
__global__ void k_gate(const float* __restrict__ ab, __bf16* __restrict__ gg) {
    const long id = (long)blockIdx.x * 256 + threadIdx.x;   // MR*FF threads
    const int j = id & (FFc - 1);
    const long row = id >> 12;
    const float a  = ab[row * (2 * FFc) + j];
    const float gt = ab[row * (2 * FFc) + FFc + j];
    const float ge = 0.5f * gt * (1.f + erff(gt * 0.70710678118654752f));
    gg[id] = (__bf16)(a * ge);
}

// ---------------------------------------------------------------------------
// Final epilogue: norm / scale / clip (+o). One block per row.
// ---------------------------------------------------------------------------
__launch_bounds__(256)
__global__ void k_final(const float* __restrict__ o, const float* __restrict__ alpha,
                        const float* __restrict__ beta, const float* __restrict__ a_s,
                        const float* __restrict__ b_s, float* __restrict__ out) {
    __shared__ float red[256];
    const int row = blockIdx.x, tid = threadIdx.x;
    float v[4]; float s = 0.f;
#pragma unroll
    for (int i = 0; i < 4; ++i) { v[i] = o[(size_t)row * Dc + tid + i * 256]; s += v[i] * v[i]; }
    red[tid] = s; __syncthreads();
    for (int st = 128; st > 0; st >>= 1) { if (tid < st) red[tid] += red[tid + st]; __syncthreads(); }
    const float ms = sqrtf(red[0]);
    const float as = fminf(fmaxf(a_s[0], 0.f), 1.f);
    const float sc = 1.f + alpha[0] * as * fminf(ms, 5.f);
    const float bb = beta[0] * b_s[0];
#pragma unroll
    for (int i = 0; i < 4; ++i) {
        int c = tid + i * 256;
        float r = v[i] * sc + bb;
        r = fminf(fmaxf(r, -10.f), 10.f) + v[i];
        out[(size_t)row * Dc + c] = r;
    }
}

// ---------------------------------------------------------------------------
// Host orchestration
// ---------------------------------------------------------------------------
extern "C" void kernel_launch(void* const* d_in, const int* in_sizes, int n_in,
                              void* d_out, int out_size, void* d_ws, size_t ws_size,
                              hipStream_t stream) {
    (void)in_sizes; (void)n_in; (void)out_size; (void)ws_size;
    const float* x     = (const float*)d_in[0];
    const float* ctx   = (const float*)d_in[1];
    const float* W_in  = (const float*)d_in[2];
    const float* b_in  = (const float*)d_in[3];
    const float* lni_g = (const float*)d_in[4];
    const float* lni_b = (const float*)d_in[5];
    const float* ln1_g = (const float*)d_in[6];
    const float* ln1_b = (const float*)d_in[7];
    const float* Wqkv  = (const float*)d_in[8];
    const float* bqkv  = (const float*)d_in[9];
    const float* Wproj = (const float*)d_in[10];
    const float* bproj = (const float*)d_in[11];
    const float* relb  = (const float*)d_in[12];
    const float* ln2_g = (const float*)d_in[13];
    const float* ln2_b = (const float*)d_in[14];
    const float* Wfc1  = (const float*)d_in[15];
    const float* bfc1  = (const float*)d_in[16];
    const float* Wfc2  = (const float*)d_in[17];
    const float* bfc2  = (const float*)d_in[18];
    const float* lnf_g = (const float*)d_in[19];
    const float* lnf_b = (const float*)d_in[20];
    const float* W_out = (const float*)d_in[21];
    const float* b_out = (const float*)d_in[22];
    const float* alpha = (const float*)d_in[23];
    const float* beta  = (const float*)d_in[24];
    const float* a_s   = (const float*)d_in[25];
    const float* b_s   = (const float*)d_in[26];

    char* ws = (char*)d_ws;
    size_t off = 0;
    auto alloc = [&](size_t bytes) -> char* {
        char* p = ws + off;
        off += (bytes + 255) & ~(size_t)255;
        return p;
    };
    float*  h    = (float*)alloc((size_t)MR * Dc * 4);
    __bf16* hn   = (__bf16*)alloc((size_t)MR * Dc * 2);
    float*  qkv  = (float*)alloc((size_t)MR * 3 * Dc * 4);
    __bf16* qb   = (__bf16*)alloc((size_t)MR * Dc * 2);
    __bf16* kb   = (__bf16*)alloc((size_t)MR * Dc * 2);
    __bf16* vb   = (__bf16*)alloc((size_t)MR * Dc * 2);
    __bf16* ao   = (__bf16*)alloc((size_t)MR * Dc * 2);
    float*  ab   = (float*)alloc((size_t)MR * 2 * FFc * 4);
    __bf16* gg   = (__bf16*)alloc((size_t)MR * FFc * 2);
    __bf16* wbuf = (__bf16*)alloc((size_t)Dc * 2 * FFc * 2);   // largest weight (8M elems)
    float*  obuf = (float*)alloc((size_t)MR * Dc * 4);

    auto f2b = [&](const float* s, __bf16* d, long n) {
        k_f2b<<<dim3((unsigned)(n / 4 / 256)), dim3(256), 0, stream>>>(s, d, n);
    };
    auto gemm = [&](const __bf16* A, const __bf16* W, const float* bias,
                    const float* res, float* C, int M, int N, int K) {
        k_gemm_bf16<<<dim3(N / 128, M / 128), dim3(256), 0, stream>>>(A, W, bias, res, C, M, N, K);
    };

    // ---- input projection + add context + LN ----
    f2b(x, hn, (long)MR * Dc);
    f2b(W_in, wbuf, (long)Dc * Dc);
    gemm(hn, wbuf, b_in, nullptr, ab /*t0 scratch*/, MR, Dc, Dc);
    k_add_ln_f32<<<dim3(MR), dim3(256), 0, stream>>>(ab, ctx, lni_g, lni_b, h);

    // ---- decoder layers ----
    for (int l = 0; l < Lc; ++l) {
        k_ln_bf16<<<dim3(MR), dim3(256), 0, stream>>>(h, ln1_g + l * Dc, ln1_b + l * Dc, hn);
        f2b(Wqkv + (size_t)l * Dc * 3 * Dc, wbuf, (long)Dc * 3 * Dc);
        gemm(hn, wbuf, bqkv + l * 3 * Dc, nullptr, qkv, MR, 3 * Dc, Dc);
        k_rope_pack<<<dim3((unsigned)((long)Bc * Tc * Hc * 32 / 256)), dim3(256), 0, stream>>>(
            qkv, qb, kb, vb);
        k_attn<<<dim3(Bc * Hc * (Tc / 128)), dim3(256), 0, stream>>>(
            qb, kb, vb, relb + (size_t)l * 257 * Hc, ao);
        f2b(Wproj + (size_t)l * Dc * Dc, wbuf, (long)Dc * Dc);
        gemm(ao, wbuf, bproj + l * Dc, h, h, MR, Dc, Dc);            // h = h + attn_out
        k_ln_bf16<<<dim3(MR), dim3(256), 0, stream>>>(h, ln2_g + l * Dc, ln2_b + l * Dc, hn);
        f2b(Wfc1 + (size_t)l * Dc * 2 * FFc, wbuf, (long)Dc * 2 * FFc);
        gemm(hn, wbuf, bfc1 + l * 2 * FFc, nullptr, ab, MR, 2 * FFc, Dc);
        k_gate<<<dim3((unsigned)((long)MR * FFc / 256)), dim3(256), 0, stream>>>(ab, gg);
        f2b(Wfc2 + (size_t)l * FFc * Dc, wbuf, (long)FFc * Dc);
        gemm(gg, wbuf, bfc2 + l * Dc, h, h, MR, Dc, FFc);            // h = h + mlp_out
    }

    // ---- final LN + output projection + epilogue ----
    k_ln_bf16<<<dim3(MR), dim3(256), 0, stream>>>(h, lnf_g, lnf_b, hn);
    f2b(W_out, wbuf, (long)Dc * Dc);
    gemm(hn, wbuf, b_out, nullptr, obuf, MR, Dc, Dc);
    k_final<<<dim3(MR), dim3(256), 0, stream>>>(obuf, alpha, beta, a_s, b_s, (float*)d_out);
}
